// RNN_Simple_83141976916376
// MI455X (gfx1250) — matrix operand order repaired and checked by reading, not compiled, not run on hardware
//
#include <hip/hip_runtime.h>
#include <hip/hip_bf16.h>

// ---------------------------------------------------------------------------
// RNN: x_proj = inputs @ W_xh + b_h (one big GEMM), then 256 serial steps of
// h = tanh(h @ W_hh + x_proj[t]).  Matmuls via v_wmma_f32_16x16x32_bf16
// (f32 accumulate).  W_hh (bf16, 8MB) stays L2-resident; h_prev is staged in
// LDS (320KB/WGP) per step via async global->LDS so A-fragments come from
// ds_load_b128.  K-loops use explicit ping-pong double buffering (no register
// copy rotation).
// ---------------------------------------------------------------------------

typedef __attribute__((ext_vector_type(16))) __bf16 v16bf;
typedef __attribute__((ext_vector_type(8)))  __bf16 v8bf;
typedef __attribute__((ext_vector_type(8)))  float  v8f;
typedef __attribute__((ext_vector_type(4)))  int    v4i;

union BF16Frag { v16bf v16; v8bf v8[2]; };

static constexpr int T_STEPS = 256;
static constexpr int BATCH   = 64;
static constexpr int D_IN    = 1024;
static constexpr int H_DIM   = 2048;

// LDS staging for the recurrence: K-chunks of h_prev [64 x KC] bf16.
// PITCH pad of +8 elems (16B) shifts consecutive rows by 4 LDS banks ->
// conflict-free ds_load_b128 across lanes.
static constexpr int KC      = 1024;
static constexpr int KC32    = KC / 32;          // wmma K-iters per chunk
static constexpr int PITCH   = KC + 8;           // elems
static constexpr int SMEM_BYTES = BATCH * PITCH * (int)sizeof(__bf16); // 132096

#if defined(__gfx1250__) && __has_builtin(__builtin_amdgcn_global_load_async_to_lds_b128)
#define USE_ASYNC_LDS 1
#else
#define USE_ASYNC_LDS 0
#endif

// ---- elementwise f32 -> bf16 ------------------------------------------------
__global__ void f32_to_bf16_kernel(const float* __restrict__ src,
                                   __bf16* __restrict__ dst, int n) {
  int i = blockIdx.x * blockDim.x + threadIdx.x;
  int stride = gridDim.x * blockDim.x;
  for (; i < n; i += stride) dst[i] = (__bf16)src[i];
}

// ---- pack row-major f32 W[K,N] into WMMA B-fragment order (bf16) -----------
// Packed layout: frag (nt,kt): 32 lanes x 16 bf16, contiguous.
// Lane l, elem e:  N = nt*16 + (l&15),  K = kt*32 + (l>>4)*8 + (e<8 ? e : e+8).
__global__ void pack_b_kernel(const float* __restrict__ W,
                              __bf16* __restrict__ Bp, int K, int N) {
  const int KT = K >> 5;
  const int NT = N >> 4;
  int idx = blockIdx.x * blockDim.x + threadIdx.x;
  if (idx >= KT * NT * 32) return;
  const int lane = idx & 31;
  const int tile = idx >> 5;          // tile = nt*KT + kt
  const int kt = tile % KT;
  const int nt = tile / KT;
  const int n = nt * 16 + (lane & 15);
  const int kbase = kt * 32 + (lane >> 4) * 8;
  __bf16* dst = Bp + ((size_t)tile * 32 + lane) * 16;
#pragma unroll
  for (int e = 0; e < 16; ++e)
    dst[e] = (__bf16)W[(size_t)(kbase + (e < 8 ? e : e + 8)) * N + n];
}

// ---- fragment loaders -------------------------------------------------------
// A (16x32 bf16) from row-major bf16 (global or LDS): two 16B runs per lane.
__device__ __forceinline__ v16bf load_a_frag(const __bf16* __restrict__ A,
                                             int lda, int m0, int k0,
                                             int lane, int half) {
  const __bf16* p = A + (size_t)(m0 + (lane & 15)) * lda + k0 + half * 8;
  BF16Frag a;
  a.v8[0] = *(const v8bf*)(p);
  a.v8[1] = *(const v8bf*)(p + 16);
  return a.v16;
}
// B (32x16 bf16) from packed buffer: one contiguous 32B per lane.
__device__ __forceinline__ v16bf load_b_frag(const __bf16* __restrict__ Bp,
                                             size_t tile, int lane) {
  const __bf16* p = Bp + (tile * 32 + lane) * 16;
  BF16Frag b;
  b.v8[0] = *(const v8bf*)(p);
  b.v8[1] = *(const v8bf*)(p + 8);
  return b.v16;
}

// ---- x_proj GEMM: [16384,1024] x [1024,2048] + bias -> d_out ---------------
// One wave = 1 M-tile x 4 N-tiles; explicit ping-pong prefetch, no reg copies.
__global__ __launch_bounds__(256) void xproj_kernel(
    const __bf16* __restrict__ Xbf,   // [T*BATCH, D_IN] row-major bf16
    const __bf16* __restrict__ Wp,    // packed W_xh fragments
    const float*  __restrict__ bias,  // [H_DIM]
    float* __restrict__ out)          // [T*BATCH, H_DIM]
{
  const int lane = threadIdx.x & 31;
  const int wave = threadIdx.x >> 5;
  const int wgid = blockIdx.x * 8 + wave;   // 0..32767
  const int mt   = wgid >> 5;               // 0..1023
  const int ng   = wgid & 31;               // group of 4 n-tiles
  const int half = lane >> 4;
  const int KT   = D_IN / 32;               // 32 (even)

  v8f acc[4] = {};
  v16bf aBuf[2];
  v16bf bBuf[2][4];
  aBuf[0] = load_a_frag(Xbf, D_IN, mt * 16, 0, lane, half);
#pragma unroll
  for (int j = 0; j < 4; ++j)
    bBuf[0][j] = load_b_frag(Wp, (size_t)(ng * 4 + j) * KT, lane);

  for (int kt = 0; kt < KT; kt += 2) {
    // even iter: compute buf0, prefetch kt+1 into buf1  (kt+1 < KT always)
    aBuf[1] = load_a_frag(Xbf, D_IN, mt * 16, (kt + 1) * 32, lane, half);
#pragma unroll
    for (int j = 0; j < 4; ++j)
      bBuf[1][j] = load_b_frag(Wp, (size_t)(ng * 4 + j) * KT + kt + 1, lane);
#pragma unroll
    for (int j = 0; j < 4; ++j)
      acc[j] = __builtin_amdgcn_wmma_f32_16x16x32_bf16(
          false, aBuf[0], false, bBuf[0][j], (short)0, acc[j], false, false);

    // odd iter: compute buf1, prefetch kt+2 into buf0 (clamped on last pass)
    const int ktn = (kt + 2 < KT) ? kt + 2 : KT - 1;
    aBuf[0] = load_a_frag(Xbf, D_IN, mt * 16, ktn * 32, lane, half);
#pragma unroll
    for (int j = 0; j < 4; ++j)
      bBuf[0][j] = load_b_frag(Wp, (size_t)(ng * 4 + j) * KT + ktn, lane);
#pragma unroll
    for (int j = 0; j < 4; ++j)
      acc[j] = __builtin_amdgcn_wmma_f32_16x16x32_bf16(
          false, aBuf[1], false, bBuf[1][j], (short)0, acc[j], false, false);
  }

#pragma unroll
  for (int j = 0; j < 4; ++j) {
    const int n = (ng * 4 + j) * 16 + (lane & 15);
    const float bv = bias[n];
#pragma unroll
    for (int v = 0; v < 8; ++v) {
      const int m = mt * 16 + v + half * 8;
      out[(size_t)m * H_DIM + n] = acc[j][v] + bv;
    }
  }
}

// ---- cooperative h_prev chunk stage into LDS --------------------------------
__device__ __forceinline__ void stage_chunk(const __bf16* __restrict__ hprev,
                                            __bf16* smem, int kc0, int tid) {
  constexpr int PIECES = (BATCH * KC) / 8;    // 16B pieces = 8192
  for (int p = tid; p < PIECES; p += 256) {   // 32 pieces/thread, uniform
    const int r = p >> 7;                      // KC/8 = 128 pieces per row
    const int c = (p & 127) * 8;
    const __bf16* gp = hprev + (size_t)r * H_DIM + kc0 + c;
    __bf16* lp = smem + r * PITCH + c;
#if USE_ASYNC_LDS
    __builtin_amdgcn_global_load_async_to_lds_b128(
        (__attribute__((address_space(1))) v4i*)gp,
        (__attribute__((address_space(3))) v4i*)lp, 0, 0);
#else
    *(v8bf*)lp = *(const v8bf*)gp;
#endif
  }
#if USE_ASYNC_LDS
#if __has_builtin(__builtin_amdgcn_s_wait_asynccnt)
  __builtin_amdgcn_s_wait_asynccnt(0);
#else
  asm volatile("s_wait_asynccnt 0x0" ::: "memory");
#endif
#endif
  __syncthreads();
}

// ---- one recurrence step: h = tanh(h_prev @ W_hh + xp) ---------------------
// 128 waves (one per N-tile), each owns the full M strip (4 tiles of 16 rows).
// A-fragments from LDS-staged h_prev; B-fragments ping-pong prefetched.
__global__ __launch_bounds__(256) void rnn_step_kernel(
    const __bf16* __restrict__ hprev,  // [BATCH, H_DIM] bf16
    const __bf16* __restrict__ Wp,     // packed W_hh fragments
    float* __restrict__ out_t,         // [BATCH, H_DIM]: xp in, h out
    __bf16* __restrict__ hnext)        // [BATCH, H_DIM] bf16
{
  extern __shared__ __bf16 smem[];     // BATCH x PITCH bf16 (132 KB)
  const int tid  = threadIdx.x;
  const int lane = tid & 31;
  const int wave = tid >> 5;
  const int nt   = blockIdx.x * 8 + wave;   // 0..127
  const int half = lane >> 4;
  const int ncol = nt * 16 + (lane & 15);
  const int KT   = H_DIM / 32;              // 64 (packed-B kt stride)

  v8f acc[4];
#pragma unroll
  for (int mt = 0; mt < 4; ++mt)
#pragma unroll
    for (int v = 0; v < 8; ++v)
      acc[mt][v] = out_t[(size_t)(mt * 16 + v + half * 8) * H_DIM + ncol];

  for (int kc = 0; kc < H_DIM; kc += KC) {
    stage_chunk(hprev, smem, kc, tid);
    const size_t base = (size_t)nt * KT + (kc >> 5);
    v16bf bBuf[2];
    bBuf[0] = load_b_frag(Wp, base, lane);
    for (int kt = 0; kt < KC32; kt += 2) {
      // even: compute with buf0, prefetch kt+1 (always in range, KC32 even)
      bBuf[1] = load_b_frag(Wp, base + kt + 1, lane);
#pragma unroll
      for (int mt = 0; mt < 4; ++mt) {
        v16bf a = load_a_frag(smem, PITCH, mt * 16, kt * 32, lane, half);
        acc[mt] = __builtin_amdgcn_wmma_f32_16x16x32_bf16(
            false, a, false, bBuf[0], (short)0, acc[mt], false, false);
      }
      // odd: compute with buf1, prefetch kt+2 (clamped on last pass)
      const int ktn = (kt + 2 < KC32) ? kt + 2 : KC32 - 1;
      bBuf[0] = load_b_frag(Wp, base + ktn, lane);
#pragma unroll
      for (int mt = 0; mt < 4; ++mt) {
        v16bf a = load_a_frag(smem, PITCH, mt * 16, (kt + 1) * 32, lane, half);
        acc[mt] = __builtin_amdgcn_wmma_f32_16x16x32_bf16(
            false, a, false, bBuf[1], (short)0, acc[mt], false, false);
      }
    }
    __syncthreads();   // protect LDS before next chunk overwrites it
  }

#pragma unroll
  for (int mt = 0; mt < 4; ++mt)
#pragma unroll
    for (int v = 0; v < 8; ++v) {
      const int m = mt * 16 + v + half * 8;
      const float y = tanhf(acc[mt][v]);
      out_t[(size_t)m * H_DIM + ncol] = y;
      hnext[(size_t)m * H_DIM + ncol] = (__bf16)y;
    }
}

// ---------------------------------------------------------------------------
extern "C" void kernel_launch(void* const* d_in, const int* in_sizes, int n_in,
                              void* d_out, int out_size, void* d_ws, size_t ws_size,
                              hipStream_t stream) {
  const float* inputs = (const float*)d_in[0];  // [T, B, D]
  const float* state  = (const float*)d_in[1];  // [B, H]
  const float* W_xh   = (const float*)d_in[2];  // [D, H]
  const float* W_hh   = (const float*)d_in[3];  // [H, H]
  const float* b_h    = (const float*)d_in[4];  // [H]
  float* out = (float*)d_out;                   // [T,B,H] then [B,H]

  // Workspace carve-up (~45 MB total)
  char* ws = (char*)d_ws;
  __bf16* Xbf = (__bf16*)(ws);                                   // 32 MiB
  __bf16* Wxp = (__bf16*)(ws + (size_t)33554432);                //  4 MiB
  __bf16* Whp = (__bf16*)(ws + (size_t)33554432 + 4194304);      //  8 MiB
  __bf16* hA  = (__bf16*)(ws + (size_t)46137344);                // 256 KiB
  __bf16* hB  = (__bf16*)(ws + (size_t)46137344 + 262144);       // 256 KiB

  // Allow >64KB dynamic LDS for the step kernel (WGP has 320KB).
  (void)hipFuncSetAttribute((const void*)rnn_step_kernel,
                            hipFuncAttributeMaxDynamicSharedMemorySize,
                            SMEM_BYTES);

  // Prep: bf16 conversions + B-fragment packing (one-time, bandwidth-bound)
  f32_to_bf16_kernel<<<4096, 256, 0, stream>>>(inputs, Xbf, T_STEPS * BATCH * D_IN);
  f32_to_bf16_kernel<<<512, 256, 0, stream>>>(state, hA, BATCH * H_DIM);
  pack_b_kernel<<<(D_IN / 32) * (H_DIM / 16) * 32 / 256, 256, 0, stream>>>(
      W_xh, Wxp, D_IN, H_DIM);
  pack_b_kernel<<<(H_DIM / 32) * (H_DIM / 16) * 32 / 256, 256, 0, stream>>>(
      W_hh, Whp, H_DIM, H_DIM);

  // x_proj + bias straight into the output buffer
  xproj_kernel<<<4096, 256, 0, stream>>>(Xbf, Wxp, b_h, out);

  // Serial recurrence: stream ordering provides the inter-step grid sync
  for (int t = 0; t < T_STEPS; ++t) {
    const __bf16* hp = (t & 1) ? hB : hA;
    __bf16*       hn = (t & 1) ? hA : hB;
    rnn_step_kernel<<<16, 256, SMEM_BYTES, stream>>>(
        hp, Whp, out + (size_t)t * BATCH * H_DIM, hn);
  }

  // final_state = outputs[T-1]
  (void)hipMemcpyAsync(out + (size_t)T_STEPS * BATCH * H_DIM,
                       out + (size_t)(T_STEPS - 1) * BATCH * H_DIM,
                       (size_t)BATCH * H_DIM * sizeof(float),
                       hipMemcpyDeviceToDevice, stream);
}
